// BiFormer_39848706573495
// MI455X (gfx1250) — compile-verified, matrix operand order checked by reading, not run on hardware
//
#include <hip/hip_runtime.h>
#include <hip/hip_bf16.h>
#include <stdint.h>

typedef __attribute__((ext_vector_type(16))) _Float16 v16h;
typedef __attribute__((ext_vector_type(8)))  _Float16 h8;
typedef __attribute__((ext_vector_type(8)))  float    v8f;

// Problem dimensions (fixed by the reference).
constexpr int Bn   = 4;
constexpr int C    = 256;
constexpr int H    = 96;
constexpr int W    = 160;
constexpr int WS   = W + 2;   // x-halo stride for conv activation buffers
constexpr int CORR = 49;      // (2*3+1)^2
constexpr int CIN0 = 305;     // 256 + 49
constexpr int C0   = 320;     // CIN0 padded to multiple of 32
constexpr int CH   = 64;
constexpr int H2   = 192;
constexpr int W2   = 320;
constexpr int HO   = 768;
constexpr int WO   = 1280;

// ---------------------------------------------------------------------------
// L2-normalize over channels: NCHW f32 -> NHWC f16 (no halo; corr checks).
// One wave per pixel.
// ---------------------------------------------------------------------------
__global__ void k_l2norm(const float* __restrict__ x, _Float16* __restrict__ out) {
  int tid = threadIdx.x, wv = tid >> 5, lane = tid & 31;
  size_t pix = (size_t)blockIdx.x * 8 + wv;           // < Bn*H*W
  int b  = (int)(pix / (H * W));
  int rm = (int)(pix % (H * W));
  int y  = rm / W, xx = rm % W;
  size_t base = (size_t)b * C * H * W + (size_t)y * W + xx;
  float v[8]; float s = 0.f;
#pragma unroll
  for (int i = 0; i < 8; ++i) {
    int c = lane + 32 * i;
    v[i] = x[base + (size_t)c * H * W];
    s += v[i] * v[i];
  }
  for (int m = 16; m; m >>= 1) s += __shfl_xor(s, m, 32);
  float sc = 1.0f / fmaxf(sqrtf(s), 1e-12f);
  _Float16* o = out + pix * C;
#pragma unroll
  for (int i = 0; i < 8; ++i) {
    int c = lane + 32 * i;
    o[c] = (_Float16)(v[i] * sc);
  }
}

// ---------------------------------------------------------------------------
// LayerNorm over channels of xt: NCHW f32 -> NHWC f16 into act0 (stride C0,
// x-halo layout), zero-fills pad channels [305,320). One wave per pixel.
// ---------------------------------------------------------------------------
__global__ void k_layernorm(const float* __restrict__ x, const float* __restrict__ g,
                            const float* __restrict__ bb, _Float16* __restrict__ out) {
  int tid = threadIdx.x, wv = tid >> 5, lane = tid & 31;
  size_t pix = (size_t)blockIdx.x * 8 + wv;
  int b  = (int)(pix / (H * W));
  int rm = (int)(pix % (H * W));
  int y  = rm / W, xx = rm % W;
  size_t base = (size_t)b * C * H * W + (size_t)y * W + xx;
  float v[8]; float s1 = 0.f, s2 = 0.f;
#pragma unroll
  for (int i = 0; i < 8; ++i) {
    int c = lane + 32 * i;
    v[i] = x[base + (size_t)c * H * W];
    s1 += v[i]; s2 += v[i] * v[i];
  }
  for (int m = 16; m; m >>= 1) { s1 += __shfl_xor(s1, m, 32); s2 += __shfl_xor(s2, m, 32); }
  float mean = s1 * (1.0f / C);
  float var  = s2 * (1.0f / C) - mean * mean;
  float inv  = rsqrtf(var + 1e-5f);
  _Float16* o = out + ((size_t)(b * H + y) * WS + xx + 1) * C0;
#pragma unroll
  for (int i = 0; i < 8; ++i) {
    int c = lane + 32 * i;
    o[c] = (_Float16)((v[i] - mean) * inv * g[c] + bb[c]);
  }
  if (lane < C0 - CIN0) o[CIN0 + lane] = (_Float16)0.f;   // zero pad channels
}

// ---------------------------------------------------------------------------
// Zero the x-halo columns (x==0 and x==WS-1) of a halo-layout NHWC buffer.
// ---------------------------------------------------------------------------
__global__ void k_zero_halo(_Float16* __restrict__ buf, int Cbuf) {
  int idx = blockIdx.x * blockDim.x + threadIdx.x;
  int total = Bn * H * 2 * Cbuf;
  if (idx >= total) return;
  int c = idx % Cbuf; int t = idx / Cbuf;
  int side = t & 1; int row = t >> 1;                  // row in [0, Bn*H)
  buf[((size_t)row * WS + (side ? (WS - 1) : 0)) * Cbuf + c] = (_Float16)0.f;
}

// ---------------------------------------------------------------------------
// Bilateral correlation: out[(dy+3)*7+(dx+3)][y][x] =
//   sum_c x1n[c, y-dy, x-dx] * x2n[c, y+dy, x+dx]  (zero outside).
// Writes into act0 (halo layout) channels [256,305).
// ---------------------------------------------------------------------------
__global__ void k_corr(const _Float16* __restrict__ x1n, const _Float16* __restrict__ x2n,
                       _Float16* __restrict__ act) {
  size_t idx = (size_t)blockIdx.x * blockDim.x + threadIdx.x;
  if (idx >= (size_t)Bn * H * W * CORR) return;
  int ch = (int)(idx % CORR);
  size_t pix = idx / CORR;
  int b  = (int)(pix / (H * W));
  int rm = (int)(pix % (H * W));
  int y  = rm / W, xx = rm % W;
  int dy = ch / 7 - 3, dx = ch % 7 - 3;
  int ya = y - dy, xa = xx - dx, yb = y + dy, xb = xx + dx;
  float s = 0.f;
  if (ya >= 0 && ya < H && xa >= 0 && xa < W && yb >= 0 && yb < H && xb >= 0 && xb < W) {
    const h8* pa = (const h8*)(x1n + ((size_t)(b * H + ya) * W + xa) * C);
    const h8* pb = (const h8*)(x2n + ((size_t)(b * H + yb) * W + xb) * C);
    for (int i = 0; i < C / 8; ++i) {
      h8 a = pa[i], bq = pb[i];
#pragma unroll
      for (int e = 0; e < 8; ++e) s += (float)a[e] * (float)bq[e];
    }
  }
  act[((size_t)(b * H + y) * WS + xx + 1) * C0 + C + ch] = (_Float16)s;
}

// ---------------------------------------------------------------------------
// Repack conv weights (Cout=64, Cin, 3, 3) f32 into per-lane WMMA B-fragment
// layout (ISA 7.12.2 16-bit layout): flat[(((tap*KC+kc)*4+nt)*32+lane)*16+e].
// ---------------------------------------------------------------------------
__global__ void k_repack(const float* __restrict__ w, int Cin, int KC,
                         _Float16* __restrict__ out) {
  int idx = blockIdx.x * blockDim.x + threadIdx.x;
  int total = 9 * KC * 4 * 512;
  if (idx >= total) return;
  int e = idx & 15; int t = idx >> 4;
  int lane = t & 31; t >>= 5;
  int nt = t & 3;   t >>= 2;
  int kc = t % KC;  int tap = t / KC;
  int hi = lane >> 4;
  int n  = nt * 16 + (lane & 15);
  int kl = (e < 8 ? e : e + 8) + hi * 8;   // K index within 32-chunk per ISA layout
  int k  = kc * 32 + kl;
  out[idx] = (k < Cin) ? (_Float16)w[(size_t)(n * Cin + k) * 9 + tap] : (_Float16)0.f;
}

// ---------------------------------------------------------------------------
// 3x3 conv via WMMA implicit GEMM.  In/out NHWC f16 with x-halo (stride WS),
// Cout=64.  One wave computes TWO 16-pixel subtiles x 64 outch (8 f32
// accumulators): each LDS B fragment feeds two WMMAs.  Thanks to the zero
// halo, A-fragment loads are unconditional global_load_b128 pairs (no exec
// masking, no zero-fill movs in the hot loop).  5 waves/block cover a full
// W=160 row => uniform tap skip; per-tap B-frag staging into LDS with b128
// copies; all 4 B frags load before the 8-WMMA burst.
// Epilogue: +bias, optional residual add (halo NHWC f16), leaky-ReLU(slope).
// ---------------------------------------------------------------------------
__global__ void __launch_bounds__(160)
k_conv3x3_wmma(const _Float16* __restrict__ in, int Cin_pad, int KC,
               const _Float16* __restrict__ wfrag, const float* __restrict__ bias,
               const _Float16* __restrict__ res, float slope,
               _Float16* __restrict__ out, int Hc) {
  extern __shared__ _Float16 lds[];
  int tid = threadIdx.x, wv = tid >> 5, lane = tid & 31;
  int m = lane & 15, hi = lane >> 4;
  int yy = (int)blockIdx.y;
  int b  = (int)blockIdx.z;
  int xbase = wv * 32;                           // wave covers 32 pixels of the row
  int stage = KC * 2048;                         // KC*4*512 halves per tap

  v8f acc[2][4] = {};
  for (int tap = 0; tap < 9; ++tap) {
    int ky = tap / 3 - 1, kx = tap % 3 - 1;
    int ys = yy + ky;
    if (ys < 0 || ys >= Hc) continue;            // uniform across block
    __syncthreads();                             // protect LDS from prev tap
    {
      const uint4* src = (const uint4*)(wfrag + (size_t)tap * stage);
      uint4* dst = (uint4*)lds;
      int n4 = stage >> 3;                       // 8 halves per uint4
      for (int i = tid; i < n4; i += (int)blockDim.x) dst[i] = src[i];
    }
    __syncthreads();

    const _Float16* rowp = in + (size_t)(b * Hc + ys) * WS * Cin_pad;
    // Prefetch next tap's activation row toward the caches (global_prefetch_b8).
    if (tap < 8) {
      int ys2 = yy + (tap + 1) / 3 - 1;
      if (ys2 >= 0 && ys2 < Hc)
        __builtin_prefetch(in + ((size_t)(b * Hc + ys2) * WS + xbase) * Cin_pad, 0, 1);
    }
    // Halo layout: interior pixel x lives at column x+1; xm in [-1, W] is
    // always in-bounds and reads zeros at the edges.
    const _Float16* p0 = rowp + (size_t)(xbase + m + kx + 1) * Cin_pad;
    const _Float16* p1 = p0 + (size_t)16 * Cin_pad;
    for (int kc = 0; kc < KC; ++kc) {
      int co = kc * 32 + hi * 8;
      v16h a0, a1;
      {
        const h8* p = (const h8*)(p0 + co);
        h8 lo = p[0], hh = p[2];
#pragma unroll
        for (int e = 0; e < 8; ++e) { a0[e] = lo[e]; a0[8 + e] = hh[e]; }
      }
      {
        const h8* p = (const h8*)(p1 + co);
        h8 lo = p[0], hh = p[2];
#pragma unroll
        for (int e = 0; e < 8; ++e) { a1[e] = lo[e]; a1[8 + e] = hh[e]; }
      }
      const v16h bf0 = *(const v16h*)&lds[((kc * 4 + 0) * 32 + lane) * 16];
      const v16h bf1 = *(const v16h*)&lds[((kc * 4 + 1) * 32 + lane) * 16];
      const v16h bf2 = *(const v16h*)&lds[((kc * 4 + 2) * 32 + lane) * 16];
      const v16h bf3 = *(const v16h*)&lds[((kc * 4 + 3) * 32 + lane) * 16];
      acc[0][0] = __builtin_amdgcn_wmma_f32_16x16x32_f16(false, a0, false, bf0, (short)0, acc[0][0], false, false);
      acc[1][0] = __builtin_amdgcn_wmma_f32_16x16x32_f16(false, a1, false, bf0, (short)0, acc[1][0], false, false);
      acc[0][1] = __builtin_amdgcn_wmma_f32_16x16x32_f16(false, a0, false, bf1, (short)0, acc[0][1], false, false);
      acc[1][1] = __builtin_amdgcn_wmma_f32_16x16x32_f16(false, a1, false, bf1, (short)0, acc[1][1], false, false);
      acc[0][2] = __builtin_amdgcn_wmma_f32_16x16x32_f16(false, a0, false, bf2, (short)0, acc[0][2], false, false);
      acc[1][2] = __builtin_amdgcn_wmma_f32_16x16x32_f16(false, a1, false, bf2, (short)0, acc[1][2], false, false);
      acc[0][3] = __builtin_amdgcn_wmma_f32_16x16x32_f16(false, a0, false, bf3, (short)0, acc[0][3], false, false);
      acc[1][3] = __builtin_amdgcn_wmma_f32_16x16x32_f16(false, a1, false, bf3, (short)0, acc[1][3], false, false);
    }
  }

  // Epilogue: D layout => lane = column n (outch), VGPR r = pixel r + 8*hi.
  int n15 = lane & 15;
#pragma unroll
  for (int s = 0; s < 2; ++s) {
    size_t pixbase = (size_t)(b * Hc + yy) * WS + xbase + s * 16 + 1;
#pragma unroll
    for (int nt = 0; nt < 4; ++nt) {
      int oc = nt * 16 + n15;
      float bv = bias[oc];
#pragma unroll
      for (int r = 0; r < 8; ++r) {
        int mm = r + hi * 8;
        float v = acc[s][nt][r] + bv;
        size_t pp = (pixbase + mm) * (size_t)CH + oc;
        if (res) v += (float)res[pp];
        v = (v >= 0.f) ? v : v * slope;
        out[pp] = (_Float16)v;
      }
    }
  }
}

// ---------------------------------------------------------------------------
// Transposed conv, stride 2, pad(1,2): out[Y][X] sums taps where (Y+ky-1)
// and (X+kx-1) are even.  wd[o][i][ky][kx] = wdec[i][o][2-ky][2-kx].
// Input y3 is halo-layout NHWC; output y4 is plain NHWC (192x320x64).
// ---------------------------------------------------------------------------
__global__ void k_deconv(const _Float16* __restrict__ y3, const float* __restrict__ wdec,
                         const float* __restrict__ bdec, _Float16* __restrict__ y4) {
  size_t idx = (size_t)blockIdx.x * blockDim.x + threadIdx.x;
  if (idx >= (size_t)Bn * H2 * W2 * CH) return;
  int o = (int)(idx % CH); size_t t = idx / CH;
  int X = (int)(t % W2); t /= W2;
  int Y = (int)(t % H2); int b = (int)(t / H2);
  float s = bdec[o];
  for (int ky = 0; ky < 3; ++ky) {
    int ay = Y + ky - 1;
    if (ay < 0 || (ay & 1)) continue;
    int yi = ay >> 1; if (yi >= H) continue;
    for (int kx = 0; kx < 3; ++kx) {
      int ax = X + kx - 1;
      if (ax < 0 || (ax & 1)) continue;
      int xi = ax >> 1; if (xi >= W) continue;
      const _Float16* ip = y3 + ((size_t)(b * H + yi) * WS + xi + 1) * CH;
      const float* wp = wdec + (size_t)o * 9 + (2 - ky) * 3 + (2 - kx);
#pragma unroll 8
      for (int i = 0; i < CH; ++i) s += (float)ip[i] * wp[(size_t)i * (CH * 9)];
    }
  }
  y4[((size_t)(b * H2 + Y) * W2 + X) * CH + o] = (_Float16)s;
}

// ---------------------------------------------------------------------------
// Final 3x3 conv 64 -> 2, output flow as NCHW f32.
// ---------------------------------------------------------------------------
__global__ void k_flow(const _Float16* __restrict__ y4, const float* __restrict__ wf,
                       const float* __restrict__ bf, float* __restrict__ flow) {
  size_t idx = (size_t)blockIdx.x * blockDim.x + threadIdx.x;
  if (idx >= (size_t)Bn * H2 * W2) return;
  int X = (int)(idx % W2); size_t t = idx / W2;
  int Y = (int)(t % H2); int b = (int)(t / H2);
  float s0 = bf[0], s1 = bf[1];
  for (int tap = 0; tap < 9; ++tap) {
    int ys = Y + tap / 3 - 1, xs = X + tap % 3 - 1;
    if (ys < 0 || ys >= H2 || xs < 0 || xs >= W2) continue;
    const _Float16* ip = y4 + ((size_t)(b * H2 + ys) * W2 + xs) * CH;
#pragma unroll 8
    for (int i = 0; i < CH; ++i) {
      float v = (float)ip[i];
      s0 += v * wf[(size_t)(0 * CH + i) * 9 + tap];
      s1 += v * wf[(size_t)(1 * CH + i) * 9 + tap];
    }
  }
  flow[((size_t)(b * 2 + 0) * H2 + Y) * W2 + X] = s0;
  flow[((size_t)(b * 2 + 1) * H2 + Y) * W2 + X] = s1;
}

// ---------------------------------------------------------------------------
// Bilinear x4 upsample (half-pixel centers, edge clamp), times 4.0.
// ---------------------------------------------------------------------------
__global__ void k_resize(const float* __restrict__ flow, float* __restrict__ out) {
  size_t idx = (size_t)blockIdx.x * blockDim.x + threadIdx.x;
  if (idx >= (size_t)Bn * 2 * HO * WO) return;
  int X = (int)(idx % WO); size_t t = idx / WO;
  int Y = (int)(t % HO); t /= HO;
  int ch = (int)(t % 2); int b = (int)(t / 2);
  float sy = (Y + 0.5f) * 0.25f - 0.5f;
  float sx = (X + 0.5f) * 0.25f - 0.5f;
  float fy0 = floorf(sy), fx0 = floorf(sx);
  float fy = sy - fy0, fx = sx - fx0;
  int y0 = min(max((int)fy0, 0), H2 - 1), y1 = min(max((int)fy0 + 1, 0), H2 - 1);
  int x0 = min(max((int)fx0, 0), W2 - 1), x1 = min(max((int)fx0 + 1, 0), W2 - 1);
  const float* F = flow + (size_t)(b * 2 + ch) * H2 * W2;
  float v = (1.f - fy) * ((1.f - fx) * F[(size_t)y0 * W2 + x0] + fx * F[(size_t)y0 * W2 + x1]) +
            fy        * ((1.f - fx) * F[(size_t)y1 * W2 + x0] + fx * F[(size_t)y1 * W2 + x1]);
  out[idx] = v * 4.0f;
}

// ---------------------------------------------------------------------------
extern "C" void kernel_launch(void* const* d_in, const int* in_sizes, int n_in,
                              void* d_out, int out_size, void* d_ws, size_t ws_size,
                              hipStream_t stream) {
  (void)in_sizes; (void)n_in; (void)out_size; (void)ws_size;
  const float* x1   = (const float*)d_in[0];
  const float* x2   = (const float*)d_in[1];
  const float* xt   = (const float*)d_in[2];
  const float* gln  = (const float*)d_in[3];
  const float* bln  = (const float*)d_in[4];
  const float* w0   = (const float*)d_in[5];
  const float* b0   = (const float*)d_in[6];
  const float* wr1  = (const float*)d_in[7];
  const float* br1  = (const float*)d_in[8];
  const float* wr2  = (const float*)d_in[9];
  const float* br2  = (const float*)d_in[10];
  const float* w2   = (const float*)d_in[11];
  const float* b2   = (const float*)d_in[12];
  const float* w3   = (const float*)d_in[13];
  const float* b3   = (const float*)d_in[14];
  const float* wdec = (const float*)d_in[15];
  const float* bdec = (const float*)d_in[16];
  const float* wf   = (const float*)d_in[17];
  const float* bf   = (const float*)d_in[18];
  // d_in[19] = corr_range (always 3; hardcoded above)
  float* out = (float*)d_out;

  char* ws = (char*)d_ws;
  size_t off = 0;
  auto take = [&](size_t bytes) { char* p = ws + off; off += (bytes + 255) & ~(size_t)255; return p; };
  _Float16* x1n  = (_Float16*)take((size_t)Bn * H * W  * C  * 2);  // 31.5 MB (no halo)
  _Float16* x2n  = (_Float16*)take((size_t)Bn * H * W  * C  * 2);  // 31.5 MB (no halo)
  _Float16* act0 = (_Float16*)take((size_t)Bn * H * WS * C0 * 2);  // halo layout
  _Float16* y0   = (_Float16*)take((size_t)Bn * H * WS * CH * 2);  // halo layout
  _Float16* buf1 = (_Float16*)take((size_t)Bn * H * WS * CH * 2);
  _Float16* buf2 = (_Float16*)take((size_t)Bn * H * WS * CH * 2);
  _Float16* wfg0 = (_Float16*)take((size_t)9 * 10 * 2048 * 2);     // conv0 frags
  _Float16* wfgA = (_Float16*)take((size_t)9 * 2 * 2048 * 2);
  _Float16* wfgB = (_Float16*)take((size_t)9 * 2 * 2048 * 2);
  _Float16* wfgC = (_Float16*)take((size_t)9 * 2 * 2048 * 2);
  _Float16* wfgD = (_Float16*)take((size_t)9 * 2 * 2048 * 2);
  _Float16* y4   = x1n;              // reuse: x1n dead after corr (same size)
  float*    flow = (float*)x2n;      // reuse: x2n dead after corr

  const int npix = Bn * H * W;       // 61440
  k_l2norm   <<<npix / 8, 256, 0, stream>>>(x1, x1n);
  k_l2norm   <<<npix / 8, 256, 0, stream>>>(x2, x2n);
  k_layernorm<<<npix / 8, 256, 0, stream>>>(xt, gln, bln, act0);
  // Zero x-halo columns of all conv-consumed buffers (ws is poisoned once).
  k_zero_halo<<<(Bn * H * 2 * C0 + 255) / 256, 256, 0, stream>>>(act0, C0);
  k_zero_halo<<<(Bn * H * 2 * CH + 255) / 256, 256, 0, stream>>>(y0,   CH);
  k_zero_halo<<<(Bn * H * 2 * CH + 255) / 256, 256, 0, stream>>>(buf1, CH);
  k_zero_halo<<<(Bn * H * 2 * CH + 255) / 256, 256, 0, stream>>>(buf2, CH);
  {
    size_t tot = (size_t)npix * CORR;
    k_corr<<<(unsigned)((tot + 255) / 256), 256, 0, stream>>>(x1n, x2n, act0);
  }
  k_repack<<<(9 * 10 * 2048 + 255) / 256, 256, 0, stream>>>(w0,  CIN0, 10, wfg0);
  k_repack<<<(9 * 2  * 2048 + 255) / 256, 256, 0, stream>>>(wr1, CH,   2,  wfgA);
  k_repack<<<(9 * 2  * 2048 + 255) / 256, 256, 0, stream>>>(wr2, CH,   2,  wfgB);
  k_repack<<<(9 * 2  * 2048 + 255) / 256, 256, 0, stream>>>(w2,  CH,   2,  wfgC);
  k_repack<<<(9 * 2  * 2048 + 255) / 256, 256, 0, stream>>>(w3,  CH,   2,  wfgD);

  dim3 cg(1, H, Bn);                 // 5 waves x 32 px = one full row per block
  k_conv3x3_wmma<<<cg, 160, 10 * 2048 * 2, stream>>>(act0, C0, 10, wfg0, b0,  nullptr, 1.0f, y0,   H);
  k_conv3x3_wmma<<<cg, 160, 2  * 2048 * 2, stream>>>(y0,   CH, 2,  wfgA, br1, nullptr, 0.1f, buf1, H);
  k_conv3x3_wmma<<<cg, 160, 2  * 2048 * 2, stream>>>(buf1, CH, 2,  wfgB, br2, y0,      0.1f, buf2, H);
  k_conv3x3_wmma<<<cg, 160, 2  * 2048 * 2, stream>>>(buf2, CH, 2,  wfgC, b2,  nullptr, 0.2f, buf1, H);
  k_conv3x3_wmma<<<cg, 160, 2  * 2048 * 2, stream>>>(buf1, CH, 2,  wfgD, b3,  nullptr, 0.2f, y0,   H);

  {
    size_t tot = (size_t)Bn * H2 * W2 * CH;
    k_deconv<<<(unsigned)(tot / 256), 256, 0, stream>>>(y0, wdec, bdec, y4);
  }
  {
    size_t tot = (size_t)Bn * H2 * W2;
    k_flow<<<(unsigned)(tot / 256), 256, 0, stream>>>(y4, wf, bf, flow);
  }
  {
    size_t tot = (size_t)Bn * 2 * HO * WO;
    k_resize<<<(unsigned)(tot / 256), 256, 0, stream>>>(flow, out);
  }
}
